// EncoderTreeMemNN_5059471474937
// MI455X (gfx1250) — matrix operand order repaired and checked by reading, not compiled, not run on hardware
//
#include <hip/hip_runtime.h>

typedef __attribute__((ext_vector_type(16))) __bf16 v16bf;
typedef __attribute__((ext_vector_type(8)))  float  v8f;

#define D     256
#define BQ    64
#define MENT  8000
#define ESZ   32
#define WVS   4      // waves per workgroup in k_main
#define EPW   8      // entries per wave in k_main

__device__ __forceinline__ unsigned short f2bf_bits(float f) {
  unsigned int u = __builtin_bit_cast(unsigned int, f);
  u += 0x7fffu + ((u >> 16) & 1u);            // round-to-nearest-even
  return (unsigned short)(u >> 16);
}
__device__ __forceinline__ __bf16 bfbits(unsigned short h) {
  return __builtin_bit_cast(__bf16, h);
}
__device__ __forceinline__ float bf2f(unsigned short h) {
  unsigned int u = ((unsigned int)h) << 16;
  return __builtin_bit_cast(float, u);
}

// ---------------------------------------------------------------------------
// Zero the atomically-accumulated output regions (o_k and combined).
__global__ void k_zero(float* __restrict__ out) {
  int i = blockIdx.x * blockDim.x + threadIdx.x;
  if (i < BQ * D) {
    out[i] = 0.f;                         // o_k
    out[BQ * D + BQ * MENT + i] = 0.f;    // combined
  }
}

// ---------------------------------------------------------------------------
// vq = query @ qproj_w^T + qproj_b ; also emit bf16 copies of query and vq.
__global__ __launch_bounds__(256) void k_project(
    const float* __restrict__ query, const float* __restrict__ w,
    const float* __restrict__ bvec, unsigned short* __restrict__ qbf,
    unsigned short* __restrict__ vqbf) {
  int b = blockIdx.x, j = threadIdx.x;
  const float* qrow = query + b * D;
  float acc = bvec[j];
  for (int k = 0; k < D; ++k) acc = fmaf(qrow[k], w[j * D + k], acc);
  vqbf[b * D + j] = f2bf_bits(acc);
  qbf[b * D + j]  = f2bf_bits(qrow[j]);
}

// ---------------------------------------------------------------------------
// logits_e[b,m] = query[b,:] . keys_e[m,:]   (bf16 WMMA, one wave per tile)
__global__ __launch_bounds__(32) void k_logits_e(
    const int* __restrict__ keys, const float* __restrict__ wordemb,
    const unsigned short* __restrict__ qbf, float* __restrict__ logits) {
  __shared__ unsigned short s_k[16 * D];
  __shared__ unsigned short s_q[16 * D];
  __shared__ int s_id[16];

  int mt = blockIdx.x % (MENT / 16);
  int bt = blockIdx.x / (MENT / 16);
  int m0 = mt * 16, b0 = bt * 16;
  int lane = threadIdx.x;
  if (lane < 16) s_id[lane] = keys[m0 + lane];
  asm volatile("s_wait_dscnt 0" ::: "memory");

  for (int i = lane; i < 16 * D; i += 32) {
    int row = i >> 8;
    int k   = i & (D - 1);
    int id  = s_id[row];
    float v = (id != 0) ? wordemb[(long)id * D + k] : 0.f;
    s_k[i] = f2bf_bits(v);
    s_q[i] = qbf[(b0 + row) * D + k];
  }
  asm volatile("s_wait_dscnt 0" ::: "memory");

  int Mr = lane & 15, hf = lane >> 4;
  v8f c = {0.f, 0.f, 0.f, 0.f, 0.f, 0.f, 0.f, 0.f};
#pragma unroll
  for (int kc = 0; kc < 8; ++kc) {
    int K0 = kc * 32;
    v16bf a, bb;
#pragma unroll
    for (int v = 0; v < 8; ++v) {
      int kof = 2 * v + 16 * hf;
      a[2 * v]     = bfbits(s_q[Mr * D + K0 + kof]);
      a[2 * v + 1] = bfbits(s_q[Mr * D + K0 + kof + 1]);
      int kb = 2 * v + 16 * hf;
      bb[2 * v]     = bfbits(s_k[Mr * D + K0 + kb]);
      bb[2 * v + 1] = bfbits(s_k[Mr * D + K0 + kb + 1]);
    }
    c = __builtin_amdgcn_wmma_f32_16x16x32_bf16(false, a, false, bb,
                                                (short)0, c, false, false);
  }
#pragma unroll
  for (int r = 0; r < 8; ++r)
    logits[(long)(b0 + r + 8 * hf) * MENT + m0 + Mr] = c[r];
}

// ---------------------------------------------------------------------------
// Per-row max and sum(exp) over m=8000 for the entry softmax.
__global__ __launch_bounds__(256) void k_rowstats(
    const float* __restrict__ logits, float* __restrict__ rmax,
    float* __restrict__ rsum) {
  __shared__ float red[256];
  int b = blockIdx.x, t = threadIdx.x;
  const float* row = logits + (long)b * MENT;
  float mx = -3.4e38f;
  for (int m = t; m < MENT; m += 256) mx = fmaxf(mx, row[m]);
  red[t] = mx; __syncthreads();
  for (int s = 128; s > 0; s >>= 1) {
    if (t < s) red[t] = fmaxf(red[t], red[t + s]);
    __syncthreads();
  }
  mx = red[0]; __syncthreads();
  float sm = 0.f;
  for (int m = t; m < MENT; m += 256) sm += __expf(row[m] - mx);
  red[t] = sm; __syncthreads();
  for (int s = 128; s > 0; s >>= 1) {
    if (t < s) red[t] += red[t + s];
    __syncthreads();
  }
  if (t == 0) { rmax[b] = mx; rsum[b] = red[0]; }
}

// ---------------------------------------------------------------------------
// Fused per-entry token-attention + entry-weighted accumulation.
__global__ __launch_bounds__(WVS * 32) void k_main(
    const int* __restrict__ keys, const int* __restrict__ entries,
    const float* __restrict__ wordemb, const unsigned short* __restrict__ vqbf,
    const float* __restrict__ logits, const float* __restrict__ rmax,
    const float* __restrict__ rsum, float* __restrict__ out_ok,
    float* __restrict__ out_comb) {
  __shared__ unsigned short s_vq[16 * D];              //  8 KB
  __shared__ unsigned short s_ents[WVS][ESZ * D];      // 64 KB
  __shared__ unsigned short s_key[WVS][D];             //  2 KB
  __shared__ float          s_pe[WVS][16];
  __shared__ unsigned short s_w[WVS][16 * ESZ];        //  4 KB
  __shared__ float          s_ok[16 * D];              // 16 KB
  __shared__ float          s_comb[16 * D];            // 16 KB

  int bt = blockIdx.x & 3;
  int chunk = blockIdx.x >> 2;
  int b0 = bt * 16;
  int tid = threadIdx.x;
  int wv = tid >> 5, lane = tid & 31;
  int Mr = lane & 15, hf = lane >> 4;

  for (int i = tid; i < 16 * D; i += WVS * 32) {
    s_ok[i] = 0.f;
    s_comb[i] = 0.f;
    s_vq[i] = vqbf[(b0 + (i >> 8)) * D + (i & (D - 1))];
  }
  __syncthreads();

  v8f Cacc[16];
#pragma unroll
  for (int nt = 0; nt < 16; ++nt)
    Cacc[nt] = (v8f){0.f, 0.f, 0.f, 0.f, 0.f, 0.f, 0.f, 0.f};

  int mbase = chunk * (WVS * EPW) + wv * EPW;

  for (int e = 0; e < EPW; ++e) {
    int m = mbase + e;

    // ---- gather entry token embeddings -> bf16 LDS slab ----
    int myid = entries[m * ESZ + lane];
    for (int tok = 0; tok < ESZ; ++tok) {
      int id = __shfl(myid, tok, 32);
      const float* src = wordemb + (long)id * D;
      for (int k = lane; k < D; k += 32)
        s_ents[wv][tok * D + k] = f2bf_bits(id != 0 ? src[k] : 0.f);
    }
    int kid = keys[m];
    const float* ksrc = wordemb + (long)kid * D;
    for (int k = lane; k < D; k += 32)
      s_key[wv][k] = f2bf_bits(kid != 0 ? ksrc[k] : 0.f);
    if (lane < 16) {
      float le = logits[(long)(b0 + lane) * MENT + m];
      s_pe[wv][lane] = __expf(le - rmax[b0 + lane]) / rsum[b0 + lane];
    }
    asm volatile("s_wait_dscnt 0" ::: "memory");

    // ---- GEMM1: logits_v = vq (16xK) x ents^T (Kx32), K=256 ----
    v8f L0 = {0.f, 0.f, 0.f, 0.f, 0.f, 0.f, 0.f, 0.f};
    v8f L1 = {0.f, 0.f, 0.f, 0.f, 0.f, 0.f, 0.f, 0.f};
#pragma unroll
    for (int kc = 0; kc < 8; ++kc) {
      int K0 = kc * 32;
      v16bf a, b0v, b1v;
#pragma unroll
      for (int v = 0; v < 8; ++v) {
        int kof = 2 * v + 16 * hf;
        a[2 * v]     = bfbits(s_vq[Mr * D + K0 + kof]);
        a[2 * v + 1] = bfbits(s_vq[Mr * D + K0 + kof + 1]);
        int kb = 2 * v + 16 * hf;
        b0v[2 * v]     = bfbits(s_ents[wv][Mr * D + K0 + kb]);
        b0v[2 * v + 1] = bfbits(s_ents[wv][Mr * D + K0 + kb + 1]);
        b1v[2 * v]     = bfbits(s_ents[wv][(16 + Mr) * D + K0 + kb]);
        b1v[2 * v + 1] = bfbits(s_ents[wv][(16 + Mr) * D + K0 + kb + 1]);
      }
      L0 = __builtin_amdgcn_wmma_f32_16x16x32_bf16(false, a, false, b0v,
                                                   (short)0, L0, false, false);
      L1 = __builtin_amdgcn_wmma_f32_16x16x32_bf16(false, a, false, b1v,
                                                   (short)0, L1, false, false);
    }

    // ---- softmax over 32 tokens; fold prob_e into the weights ----
#pragma unroll
    for (int r = 0; r < 8; ++r) {
      float x0 = L0[r], x1 = L1[r];
      float mx = fmaxf(x0, x1);
      mx = fmaxf(mx, __shfl_xor(mx, 1, 32));
      mx = fmaxf(mx, __shfl_xor(mx, 2, 32));
      mx = fmaxf(mx, __shfl_xor(mx, 4, 32));
      mx = fmaxf(mx, __shfl_xor(mx, 8, 32));
      float e0 = __expf(x0 - mx), e1 = __expf(x1 - mx);
      float s = e0 + e1;
      s += __shfl_xor(s, 1, 32);
      s += __shfl_xor(s, 2, 32);
      s += __shfl_xor(s, 4, 32);
      s += __shfl_xor(s, 8, 32);
      int row = r + 8 * hf;
      float pw = s_pe[wv][row] / s;
      s_w[wv][row * ESZ + Mr]      = f2bf_bits(e0 * pw);
      s_w[wv][row * ESZ + 16 + Mr] = f2bf_bits(e1 * pw);
    }
    asm volatile("s_wait_dscnt 0" ::: "memory");

    // ---- GEMM2: combined += w (16x32) x ents (32x256) ----
    v16bf a2;
#pragma unroll
    for (int v = 0; v < 8; ++v) {
      int kof = 2 * v + 16 * hf;
      a2[2 * v]     = bfbits(s_w[wv][Mr * ESZ + kof]);
      a2[2 * v + 1] = bfbits(s_w[wv][Mr * ESZ + kof + 1]);
    }
#pragma unroll
    for (int nt = 0; nt < 16; ++nt) {
      v16bf bv;
#pragma unroll
      for (int v = 0; v < 8; ++v) {
        int kb = 2 * v + 16 * hf;  // token index (K)
        bv[2 * v]     = bfbits(s_ents[wv][kb * D + nt * 16 + Mr]);
        bv[2 * v + 1] = bfbits(s_ents[wv][(kb + 1) * D + nt * 16 + Mr]);
      }
      Cacc[nt] = __builtin_amdgcn_wmma_f32_16x16x32_bf16(
          false, a2, false, bv, (short)0, Cacc[nt], false, false);
    }

    // ---- o_k += prob_e[b,m] * keys_e[m,:] (ds_add_f32) ----
    for (int i = lane; i < 16 * D; i += 32) {
      int row = i >> 8, k = i & (D - 1);
      atomicAdd(&s_ok[i], s_pe[wv][row] * bf2f(s_key[wv][k]));
    }
  }

  // flush register accumulators into shared, then global atomics
#pragma unroll
  for (int nt = 0; nt < 16; ++nt)
#pragma unroll
    for (int r = 0; r < 8; ++r)
      atomicAdd(&s_comb[(r + 8 * hf) * D + nt * 16 + Mr], Cacc[nt][r]);
  __syncthreads();
  for (int i = tid; i < 16 * D; i += WVS * 32) {
    int row = i >> 8, k = i & (D - 1);
    atomicAdd(&out_ok[(b0 + row) * D + k], s_ok[i]);
    atomicAdd(&out_comb[(b0 + row) * D + k], s_comb[i]);
  }
}

// ---------------------------------------------------------------------------
extern "C" void kernel_launch(void* const* d_in, const int* in_sizes, int n_in,
                              void* d_out, int out_size, void* d_ws,
                              size_t ws_size, hipStream_t stream) {
  (void)in_sizes; (void)n_in; (void)out_size; (void)ws_size;
  const int*   keys    = (const int*)d_in[0];
  const int*   entries = (const int*)d_in[1];
  const float* query   = (const float*)d_in[2];
  const float* wordemb = (const float*)d_in[3];
  const float* qw      = (const float*)d_in[4];
  const float* qb      = (const float*)d_in[5];

  float* out      = (float*)d_out;
  float* out_ok   = out;                       // (64,256)
  float* out_log  = out + BQ * D;              // (64,8000)
  float* out_comb = out + BQ * D + BQ * MENT;  // (64,256)

  unsigned short* qbf  = (unsigned short*)d_ws;
  unsigned short* vqbf = qbf + BQ * D;
  float* rmax = (float*)(vqbf + BQ * D);
  float* rsum = rmax + BQ;

  k_zero<<<(BQ * D + 255) / 256, 256, 0, stream>>>(out);
  k_project<<<BQ, D, 0, stream>>>(query, qw, qb, qbf, vqbf);
  k_logits_e<<<(MENT / 16) * (BQ / 16), 32, 0, stream>>>(keys, wordemb, qbf,
                                                         out_log);
  k_rowstats<<<BQ, 256, 0, stream>>>(out_log, rmax, rsum);
  k_main<<<(BQ / 16) * (MENT / (WVS * EPW)), WVS * 32, 0, stream>>>(
      keys, entries, wordemb, vqbf, out_log, rmax, rsum, out_ok, out_comb);
}
